// Self_Attn_86766929313863
// MI455X (gfx1250) — matrix operand order repaired and checked by reading, not compiled, not run on hardware
//
#include <hip/hip_runtime.h>
#include <hip/hip_bf16.h>

typedef unsigned short u16;
typedef unsigned int   u32;

typedef __attribute__((ext_vector_type(16))) __bf16 v16bf;
typedef __attribute__((ext_vector_type(8)))  __bf16 v8bf;
typedef __attribute__((ext_vector_type(8)))  float  v8f;

#define BB  8
#define SS  2048
#define HH  1024
#define DQK 128
#define SCP 132   // padded LDS stride for score buffer (bank-conflict avoidance)

__device__ __forceinline__ u16 f2bf(float f) {
  union { float f; u32 u; } v; v.f = f;
  u32 r = v.u + 0x7FFFu + ((v.u >> 16) & 1u);   // round-to-nearest-even
  return (u16)(r >> 16);
}

__device__ __forceinline__ v16bf cat8(v8bf lo, v8bf hi) {
  return __builtin_shufflevector(lo, hi, 0,1,2,3,4,5,6,7,8,9,10,11,12,13,14,15);
}

__device__ __forceinline__ v8f wmma_bf16(v16bf a, v16bf b, v8f c) {
  // D = A(16x32 bf16) * B(32x16 bf16) + C(16x16 f32)
  return __builtin_amdgcn_wmma_f32_16x16x32_bf16(false, a, false, b, (short)0, c, false, false);
}

// A-fragment loader: 16x32 A in contiguous-k order (B loads use the same order). lane (0-15|16-31) holds row M=lane%16;
// K = kbase+{0..15} in elems 0..15, kbase = (lane<16)?0:16.
__device__ __forceinline__ v16bf load_afrag(const u16* rowptr, int c, int hb) {
  const int k0 = c * 32 + hb * 16;
  return cat8(*(const v8bf*)(rowptr + k0), *(const v8bf*)(rowptr + k0 + 8));
}

// ---------------------------------------------------------------------------
// x (f32) -> bf16, 4 elements/thread
__global__ __launch_bounds__(256) void cvt_x(const float* __restrict__ in,
                                             u16* __restrict__ o, int n4) {
  int i = blockIdx.x * 256 + threadIdx.x;
  if (i >= n4) return;
  float4 v = ((const float4*)in)[i];
  uint2 p;
  p.x = (u32)f2bf(v.x) | ((u32)f2bf(v.y) << 16);
  p.y = (u32)f2bf(v.z) | ((u32)f2bf(v.w) << 16);
  ((uint2*)o)[i] = p;
}

// W [K=1024][N] f32 -> WT [N][1024] bf16
__global__ __launch_bounds__(256) void cvt_T(const float* __restrict__ W,
                                             u16* __restrict__ WT, int N) {
  int i = blockIdx.x * 256 + threadIdx.x;
  if (i >= 1024 * N) return;
  int k = i / N, n = i % N;
  WT[(size_t)n * 1024 + k] = f2bf(W[i]);
}

// ---------------------------------------------------------------------------
// C[M,N] = A[M,1024] * WT[N,1024]^T + bias.  Block: 16 rows x 128 cols, wave = 16 cols.
// mode 0: store bf16 row-major [M][N]   (q, k)
// mode 1: store bf16 transposed vT[b][h][s]  (v)
__global__ __launch_bounds__(256) void gemm_bf16(const u16* __restrict__ A,
                                                 const u16* __restrict__ WT,
                                                 const float* __restrict__ bias,
                                                 u16* __restrict__ o,
                                                 int N, int mode) {
  const int tid = threadIdx.x;
  const int wv = tid >> 5, lane = tid & 31, l16 = lane & 15, hb = lane >> 4;
  const int m0  = blockIdx.x * 16;
  const int col = blockIdx.y * 128 + wv * 16 + l16;

  const u16* ar = A  + (size_t)(m0 + l16) * 1024;
  const u16* br = WT + (size_t)col * 1024;          // B-frag: contiguous K per lane

  v8f acc = {0.f,0.f,0.f,0.f,0.f,0.f,0.f,0.f};
  #pragma unroll 4
  for (int kk = 0; kk < 32; ++kk) {
    v16bf af = load_afrag(ar, kk, hb);
    v16bf bf = *(const v16bf*)(br + kk * 32 + hb * 16);
    acc = wmma_bf16(af, bf, acc);
  }

  const float bs = bias[col];
  if (mode == 0) {
    #pragma unroll
    for (int r = 0; r < 8; ++r) {
      const int row = m0 + r + hb * 8;            // C layout: VGPR r -> M = r + 8*(lane>=16)
      o[(size_t)row * N + col] = f2bf(acc[r] + bs);
    }
  } else {
    const int b  = m0 >> 11;                      // 2048 rows per batch
    const int s0 = (m0 & 2047) + hb * 8;          // 8 consecutive s values per lane
    u16* dst = o + (size_t)b * HH * SS + (size_t)col * SS + s0;
    uint4 w;
    w.x = (u32)f2bf(acc[0] + bs) | ((u32)f2bf(acc[1] + bs) << 16);
    w.y = (u32)f2bf(acc[2] + bs) | ((u32)f2bf(acc[3] + bs) << 16);
    w.z = (u32)f2bf(acc[4] + bs) | ((u32)f2bf(acc[5] + bs) << 16);
    w.w = (u32)f2bf(acc[6] + bs) | ((u32)f2bf(acc[7] + bs) << 16);
    *(uint4*)dst = w;                             // 16B aligned, s-contiguous
  }
}

// ---------------------------------------------------------------------------
// Flash attention: block = 32 query rows (2 M-subtiles), 8 waves.
// Wave w: score tiles for keys [kb*128+w*16, +16), then P.V for output columns
// [w*128, +128). Every K/V B-fragment is reused for both M-subtiles (2x
// arithmetic intensity vs 16-row tiles -> L2 traffic halved to ~2.3 GB).
__global__ __launch_bounds__(256) void attn_fwd(const u16* __restrict__ qb,
                                                const u16* __restrict__ kb,
                                                const u16* __restrict__ vT,
                                                const float* __restrict__ x,
                                                const float* __restrict__ gamma,
                                                float* __restrict__ out) {
  __shared__ float sc[32 * SCP];      // raw scores, padded stride
  __shared__ u16   pbuf[32 * 128];    // exp(s - m) in bf16 (A-matrix for P.V)
  __shared__ float mrow[32], lrow[32], arow[32];

  const int tid = threadIdx.x;
  const int wv = tid >> 5, lane = tid & 31, l16 = lane & 15, hb = lane >> 4;
  const int b = blockIdx.x >> 6;            // 64 tiles of 32 rows per batch
  const int qbase = (blockIdx.x & 63) << 5;

  if (tid < 32) { mrow[tid] = -3.0e38f; lrow[tid] = 0.f; }

  v8f acc[2][8];
  #pragma unroll
  for (int mt = 0; mt < 2; ++mt)
    #pragma unroll
    for (int t = 0; t < 8; ++t) acc[mt][t] = (v8f){0.f,0.f,0.f,0.f,0.f,0.f,0.f,0.f};

  const u16* qrow0 = qb + (size_t)(b * SS + qbase + l16) * DQK;
  const u16* qrow1 = qrow0 + (size_t)16 * DQK;

  for (int kbk = 0; kbk < 16; ++kbk) {
    // ---- phase 1: S = Q K^T ; one K B-fragment feeds both M-subtiles
    {
      const int key0 = kbk * 128 + wv * 16;
      const u16* krow = kb + (size_t)(b * SS + key0 + l16) * DQK;  // contiguous d
      v8f s0 = (v8f){0.f,0.f,0.f,0.f,0.f,0.f,0.f,0.f};
      v8f s1 = (v8f){0.f,0.f,0.f,0.f,0.f,0.f,0.f,0.f};
      #pragma unroll
      for (int c = 0; c < 4; ++c) {
        v16bf bf = *(const v16bf*)(krow + c * 32 + hb * 16);
        v16bf a0 = load_afrag(qrow0, c, hb);   // q tile is L0-resident, cheap reload
        v16bf a1 = load_afrag(qrow1, c, hb);
        s0 = wmma_bf16(a0, bf, s0);
        s1 = wmma_bf16(a1, bf, s1);
      }
      #pragma unroll
      for (int r = 0; r < 8; ++r) {
        sc[(r + hb * 8) * SCP + wv * 16 + l16]        = s0[r];
        sc[(16 + r + hb * 8) * SCP + wv * 16 + l16]   = s1[r];
      }
    }
    __syncthreads();  // B1: scores complete

    // ---- phase 2a: per-row running max + rescale factor (wave 0, lane-per-row)
    if (tid < 32) {
      float mx = -3.0e38f;
      #pragma unroll 8
      for (int j = 0; j < 128; ++j) mx = fmaxf(mx, sc[tid * SCP + j]);
      const float mold = mrow[tid];
      const float mnew = fmaxf(mold, mx);
      const float al = __expf(mold - mnew);
      arow[tid] = al;
      mrow[tid] = mnew;
      lrow[tid] *= al;
    }
    __syncthreads();  // B2: m/alpha ready

    // ---- phase 2b: p = exp(s - m) -> bf16, partial denominators; rescale acc
    {
      const int row = tid >> 3;           // 32 rows x 8 threads
      const int j0 = (tid & 7) * 16;      // 16 elements per thread
      const float mn = mrow[row];
      float ps = 0.f;
      #pragma unroll
      for (int jj = 0; jj < 16; ++jj) {
        const float p = __expf(sc[row * SCP + j0 + jj] - mn);
        ps += p;
        pbuf[row * 128 + j0 + jj] = f2bf(p);
      }
      atomicAdd(&lrow[row], ps);    // ds_add_f32
    }
    #pragma unroll
    for (int mt = 0; mt < 2; ++mt)
      #pragma unroll
      for (int r = 0; r < 8; ++r) {
        const float al = arow[mt * 16 + r + hb * 8];
        #pragma unroll
        for (int t = 0; t < 8; ++t) acc[mt][t][r] *= al;
      }
    __syncthreads();  // B3: pbuf complete

    // ---- phase 3: acc += P (32x128) . V (128 x 128-slice); V frag reused 2x
    v16bf pa[2][4];
    #pragma unroll
    for (int mt = 0; mt < 2; ++mt) {
      const u16* prow = pbuf + (mt * 16 + l16) * 128;
      #pragma unroll
      for (int c = 0; c < 4; ++c) pa[mt][c] = load_afrag(prow, c, hb);  // ds_load_b128
    }
    #pragma unroll
    for (int nt = 0; nt < 8; ++nt) {
      const int h = wv * 128 + nt * 16 + l16;
      const u16* vcol = vT + (size_t)b * HH * SS + (size_t)h * SS + kbk * 128;
      #pragma unroll
      for (int c = 0; c < 4; ++c) {
        v16bf bf = *(const v16bf*)(vcol + c * 32 + hb * 16);  // contiguous s
        acc[0][nt] = wmma_bf16(pa[0][c], bf, acc[0][nt]);
        acc[1][nt] = wmma_bf16(pa[1][c], bf, acc[1][nt]);
      }
    }
    // no barrier needed here: next-iter LDS writes are ordered by B1/B2 arrival
  }

  // ---- epilogue: out = gamma * (acc / l) + x
  const float g = *gamma;
  float linv[2][8];
  #pragma unroll
  for (int mt = 0; mt < 2; ++mt)
    #pragma unroll
    for (int r = 0; r < 8; ++r) linv[mt][r] = 1.0f / lrow[mt * 16 + r + hb * 8];
  #pragma unroll
  for (int mt = 0; mt < 2; ++mt)
    #pragma unroll
    for (int nt = 0; nt < 8; ++nt) {
      const int col = wv * 128 + nt * 16 + l16;
      #pragma unroll
      for (int r = 0; r < 8; ++r) {
        const int row = qbase + mt * 16 + r + hb * 8;
        const size_t idx = (size_t)(b * SS + row) * HH + col;
        out[idx] = g * (acc[mt][nt][r] * linv[mt][r]) + x[idx];
      }
    }
}

// ---------------------------------------------------------------------------
extern "C" void kernel_launch(void* const* d_in, const int* in_sizes, int n_in,
                              void* d_out, int out_size, void* d_ws, size_t ws_size,
                              hipStream_t stream) {
  const float* x     = (const float*)d_in[0];
  const float* Wq    = (const float*)d_in[1];
  const float* bq    = (const float*)d_in[2];
  const float* Wk    = (const float*)d_in[3];
  const float* bk    = (const float*)d_in[4];
  const float* Wv    = (const float*)d_in[5];
  const float* bv    = (const float*)d_in[6];
  const float* gamma = (const float*)d_in[7];
  float* out = (float*)d_out;

  // workspace partition (bf16 elements), ~78 MB total
  u16* xb  = (u16*)d_ws;                      // [16384][1024]
  u16* WqT = xb  + (size_t)16384 * 1024;      // [128][1024]
  u16* WkT = WqT + (size_t)128 * 1024;        // [128][1024]
  u16* WvT = WkT + (size_t)128 * 1024;        // [1024][1024]
  u16* qb  = WvT + (size_t)1024 * 1024;       // [16384][128]
  u16* kbf = qb  + (size_t)16384 * 128;       // [16384][128]
  u16* vT  = kbf + (size_t)16384 * 128;       // [8][1024][2048]

  cvt_x<<<16384, 256, 0, stream>>>(x, xb, 16384 * 1024 / 4);
  cvt_T<<<512,  256, 0, stream>>>(Wq, WqT, 128);
  cvt_T<<<512,  256, 0, stream>>>(Wk, WkT, 128);
  cvt_T<<<4096, 256, 0, stream>>>(Wv, WvT, 1024);

  gemm_bf16<<<dim3(1024, 1), 256, 0, stream>>>(xb, WqT, bq, qb,  128,  0);
  gemm_bf16<<<dim3(1024, 1), 256, 0, stream>>>(xb, WkT, bk, kbf, 128,  0);
  gemm_bf16<<<dim3(1024, 8), 256, 0, stream>>>(xb, WvT, bv, vT,  1024, 1);

  attn_fwd<<<512, 256, 0, stream>>>(qb, kbf, vT, x, gamma, out);
}
